// FeaturePropagation_22531398435369
// MI455X (gfx1250) — matrix operand order repaired and checked by reading, not compiled, not run on hardware
//
#include <hip/hip_runtime.h>

// Problem constants (from reference)
#define BB   16
#define NN1  4096
#define NN2  1024
#define CC1  128
#define CC2  256
#define DIN  384      // CC2 + CC1
#define NOUT 256
#define ROWS (BB * NN1)   // 65536

typedef __attribute__((ext_vector_type(16))) __bf16 v16bf;
typedef __attribute__((ext_vector_type(8)))  float  v8f;

union AFrag { v16bf v; int4 q[2]; };

// ---------------------------------------------------------------------------
// Kernel 0: shuffle f32 weights [K,N] row-major into WMMA B-fragment layout
// (bf16). Fragment f = (ntile*KT + ktile); per lane 16 contiguous bf16:
//   lane<16 : N = nt*16+lane,    K = kt*32 + 0..15
//   lane>=16: N = nt*16+lane-16, K = kt*32 + 16..31
// ---------------------------------------------------------------------------
__global__ __launch_bounds__(256) void fp_shuffle_w(
    const float* __restrict__ W, __bf16* __restrict__ out, int K, int N) {
  int t = blockIdx.x * blockDim.x + threadIdx.x;
  int lane = t & 31;
  int f = t >> 5;
  int KT = K >> 5;
  int NT = N >> 4;
  if (f >= KT * NT) return;
  int nt = f / KT, kt = f % KT;
  int ncol = nt * 16 + (lane & 15);
  int kb = kt * 32 + ((lane >> 4) ? 16 : 0);
  __bf16* dst = out + ((size_t)(f * 32 + lane)) * 16;
#pragma unroll
  for (int j = 0; j < 16; ++j)
    dst[j] = (__bf16)W[(size_t)(kb + j) * N + ncol];
}

// ---------------------------------------------------------------------------
// Kernel 1: per query point, 3-NN over 1024 ref points (xyz2 staged in LDS),
// inverse-distance weights, interpolate feat2, concat feat1, emit bf16 row.
// One wave (32 lanes) per query point, 8 points per 256-thread block.
// ---------------------------------------------------------------------------
__global__ __launch_bounds__(256) void fp_knn_interp(
    const float* __restrict__ xyz1, const float* __restrict__ feat1,
    const float* __restrict__ xyz2, const float* __restrict__ feat2,
    __bf16* __restrict__ X) {
  __shared__ float sx[NN2 * 3];
  const int tid  = threadIdx.x;
  const int wave = tid >> 5;
  const int lane = tid & 31;
  const int b = blockIdx.x >> 9;                 // 512 blocks per batch
  const int n = ((blockIdx.x & 511) << 3) + wave;

  // stage xyz2[b] (12 KB) into LDS
  const float* src = xyz2 + (size_t)b * NN2 * 3;
  for (int i = tid; i < NN2 * 3; i += 256) sx[i] = src[i];
  __syncthreads();

  const float* p1 = xyz1 + ((size_t)b * NN1 + n) * 3;
  const float px = p1[0], py = p1[1], pz = p1[2];

  float d0 = 1e30f, d1 = 1e30f, d2 = 1e30f;
  int   i0 = 0,     i1 = 0,     i2 = 0;
  auto ins = [&](float e, int j) {
    if (e < d0)      { d2 = d1; i2 = i1; d1 = d0; i1 = i0; d0 = e; i0 = j; }
    else if (e < d1) { d2 = d1; i2 = i1; d1 = e;  i1 = j; }
    else if (e < d2) { d2 = e;  i2 = j; }
  };

  // each lane scans 32 candidates, stride-32 for LDS bank spread
#pragma unroll 4
  for (int j = 0; j < 32; ++j) {
    int p = lane + (j << 5);
    float dx = px - sx[p * 3 + 0];
    float dy = py - sx[p * 3 + 1];
    float dz = pz - sx[p * 3 + 2];
    ins(dx * dx + dy * dy + dz * dz, p);
  }
  // butterfly merge of per-lane top-3 lists (wave32)
  for (int off = 16; off > 0; off >>= 1) {
    float e0 = __shfl_xor(d0, off, 32), e1 = __shfl_xor(d1, off, 32),
          e2 = __shfl_xor(d2, off, 32);
    int   j0 = __shfl_xor(i0, off, 32), j1 = __shfl_xor(i1, off, 32),
          j2 = __shfl_xor(i2, off, 32);
    ins(e0, j0); ins(e1, j1); ins(e2, j2);
  }

  float w0 = 1.f / fmaxf(d0, 1e-10f);
  float w1 = 1.f / fmaxf(d1, 1e-10f);
  float w2 = 1.f / fmaxf(d2, 1e-10f);
  float wsum = w0 + w1 + w2;
  w0 /= wsum; w1 /= wsum; w2 /= wsum;

  const float* f0 = feat2 + ((size_t)b * NN2 + i0) * CC2;
  const float* f1 = feat2 + ((size_t)b * NN2 + i1) * CC2;
  const float* f2 = feat2 + ((size_t)b * NN2 + i2) * CC2;
  const size_t row = (size_t)b * NN1 + n;
  __bf16* xr = X + row * DIN;
#pragma unroll
  for (int j = 0; j < 8; ++j) {           // interpolated CC2 channels
    int c = lane + (j << 5);
    xr[c] = (__bf16)(w0 * f0[c] + w1 * f1[c] + w2 * f2[c]);
  }
  const float* fr = feat1 + row * CC1;
#pragma unroll
  for (int j = 0; j < 4; ++j) {           // skip CC1 channels
    int c = lane + (j << 5);
    xr[CC2 + c] = (__bf16)fr[c];
  }
}

// ---------------------------------------------------------------------------
// Kernel 2/3: bf16 WMMA GEMM + bias + ReLU.  out = relu(A[M,K] @ W[K,256] + b)
//
// Block = 8 waves sharing ONE 64-column group; each wave owns a 16-row mtile.
// The block's weight fragments (4 ntiles x KT ksteps, 32-48 KB) are staged
// into LDS once (8x amortization of global weight traffic), interleaved as
// [frame][chunk16][lane] so ds_load_b128 reads have 16B lane stride (<=2-way
// bank conflict) instead of 32B (4-way).  Inner loop: 2 global b128 A loads
// (double-buffered, streaming from HBM) + 8 ds loads feed 4 WMMAs per kstep.
// ---------------------------------------------------------------------------
template <bool OUT_BF16, int K>
__global__ __launch_bounds__(256) void fp_gemm_bias_relu(
    const __bf16* __restrict__ A, const __bf16* __restrict__ Bf,
    const float* __restrict__ bias, void* __restrict__ outp, int M) {
  constexpr int KT = K / 32;
  constexpr int FRAMES = 4 * KT;                 // (ntile, kstep) frames
  __shared__ int4 sB[FRAMES * 64];               // 64 x 16B chunks per frame

  const int tid    = threadIdx.x;
  const int lane   = tid & 31;
  const int wave   = tid >> 5;
  const int mblock = blockIdx.x >> 2;
  const int ngroup = blockIdx.x & 3;

  // ---- stage this ngroup's B fragments into LDS, chunk-interleaved --------
  // src chunk i (16B) of frame fk: lane_s = (i&63)>>1, j = i&1
  // dst chunk  = fk*64 + j*32 + lane_s
  {
    const int4* src = (const int4*)(Bf + (size_t)ngroup * 4 * KT * 512);
#pragma unroll
    for (int i = tid; i < FRAMES * 64; i += 256) {
      int fk = i >> 6, r = i & 63;
      sB[fk * 64 + (r & 1) * 32 + (r >> 1)] = src[i];
    }
  }
  __syncthreads();

  const int mbase = (mblock * 8 + wave) << 4;
  if (mbase >= M) return;

  // A fragment: lane<16 -> row mbase+lane, K 0..15;
  //             lane>=16 -> row mbase+lane-16, K 16..31
  const int arow = mbase + (lane & 15);
  const int asub = (lane >> 4) << 4;             // 0 or 16 (bf16 elems)
  const __bf16* arp = A + (size_t)arow * K + asub;

  auto loadA = [&](AFrag& a, int kt) {
    const __bf16* p = arp + kt * 32;
    a.q[0] = *(const int4*)(p);
    a.q[1] = *(const int4*)(p + 8);
  };
  auto loadB = [&](int t, int kt) -> v16bf {
    AFrag b;
    int c0 = (t * KT + kt) * 64 + lane;          // chunk j=0
    b.q[0] = sB[c0];
    b.q[1] = sB[c0 + 32];                        // chunk j=1
    return b.v;
  };

  v8f acc0 = {}, acc1 = {}, acc2 = {}, acc3 = {};
  AFrag aBuf0, aBuf1;
  loadA(aBuf0, 0);
#pragma unroll
  for (int kt = 0; kt < KT; ++kt) {
    AFrag& ac = (kt & 1) ? aBuf1 : aBuf0;
    AFrag& an = (kt & 1) ? aBuf0 : aBuf1;
    if (kt + 1 < KT) loadA(an, kt + 1);          // prefetch next A (global)
    v16bf b0 = loadB(0, kt);
    v16bf b1 = loadB(1, kt);
    v16bf b2 = loadB(2, kt);
    v16bf b3 = loadB(3, kt);
    acc0 = __builtin_amdgcn_wmma_f32_16x16x32_bf16(false, ac.v, false, b0,
                                                   (short)0, acc0, false, false);
    acc1 = __builtin_amdgcn_wmma_f32_16x16x32_bf16(false, ac.v, false, b1,
                                                   (short)0, acc1, false, false);
    acc2 = __builtin_amdgcn_wmma_f32_16x16x32_bf16(false, ac.v, false, b2,
                                                   (short)0, acc2, false, false);
    acc3 = __builtin_amdgcn_wmma_f32_16x16x32_bf16(false, ac.v, false, b3,
                                                   (short)0, acc3, false, false);
  }

  // C/D layout: lanes 0-15 -> N=lane, M=r ; lanes 16-31 -> N=lane-16, M=8+r
  const int mo = mbase + ((lane >> 4) << 3);
#define FP_EPILOGUE(ACC, T)                                                    \
  {                                                                            \
    int nc = (ngroup << 6) + ((T) << 4) + (lane & 15);                         \
    float bb = bias[nc];                                                       \
    _Pragma("unroll") for (int r = 0; r < 8; ++r) {                            \
      float v = fmaxf((ACC)[r] + bb, 0.0f);                                    \
      size_t o = (size_t)(mo + r) * NOUT + nc;                                 \
      if constexpr (OUT_BF16) ((__bf16*)outp)[o] = (__bf16)v;                  \
      else                    ((float*)outp)[o] = v;                           \
    }                                                                          \
  }
  FP_EPILOGUE(acc0, 0)
  FP_EPILOGUE(acc1, 1)
  FP_EPILOGUE(acc2, 2)
  FP_EPILOGUE(acc3, 3)
#undef FP_EPILOGUE
}

// ---------------------------------------------------------------------------
extern "C" void kernel_launch(void* const* d_in, const int* in_sizes, int n_in,
                              void* d_out, int out_size, void* d_ws, size_t ws_size,
                              hipStream_t stream) {
  const float* xyz1  = (const float*)d_in[0];
  const float* feat1 = (const float*)d_in[1];
  const float* xyz2  = (const float*)d_in[2];
  const float* feat2 = (const float*)d_in[3];
  const float* W1    = (const float*)d_in[4];
  const float* b1    = (const float*)d_in[5];
  const float* W2    = (const float*)d_in[6];
  const float* b2    = (const float*)d_in[7];

  char* ws = (char*)d_ws;
  const size_t xBytes   = (size_t)ROWS * DIN * 2;   // 50,331,648
  const size_t hBytes   = (size_t)ROWS * NOUT * 2;  // 33,554,432
  const size_t w1fBytes = (size_t)DIN * NOUT * 2;   // 196,608
  __bf16* X   = (__bf16*)ws;
  __bf16* H   = (__bf16*)(ws + xBytes);
  __bf16* W1f = (__bf16*)(ws + xBytes + hBytes);
  __bf16* W2f = (__bf16*)(ws + xBytes + hBytes + w1fBytes);

  // Stage 0: weight shuffles (tiny)
  fp_shuffle_w<<<24, 256, 0, stream>>>(W1, W1f, DIN,  NOUT);  // 192 frags
  fp_shuffle_w<<<16, 256, 0, stream>>>(W2, W2f, NOUT, NOUT);  // 128 frags

  // Stage 1: 3-NN + interpolation + concat -> X bf16 [65536 x 384]
  fp_knn_interp<<<ROWS / 8, 256, 0, stream>>>(xyz1, feat1, xyz2, feat2, X);

  // Stage 2: H = relu(X @ W1 + b1)  (bf16 out)
  // grid: (ROWS/128) mblocks x 4 ngroups
  fp_gemm_bias_relu<true, DIN><<<(ROWS / 128) * 4, 256, 0, stream>>>(
      X, W1f, b1, H, ROWS);

  // Stage 3: out = relu(H @ W2 + b2)  (f32 out)
  fp_gemm_bias_relu<false, NOUT><<<(ROWS / 128) * 4, 256, 0, stream>>>(
      H, W2f, b2, d_out, ROWS);
}
